// RecruitmentLayer_14173392077104
// MI455X (gfx1250) — matrix operand order repaired and checked by reading, compiled, not run on hardware
//
#include <hip/hip_runtime.h>
#include <hip/hip_bf16.h>

typedef __attribute__((ext_vector_type(16))) _Float16 v16h;
typedef __attribute__((ext_vector_type(8)))  _Float16 v8h;
typedef __attribute__((ext_vector_type(4)))  _Float16 v4h;
typedef __attribute__((ext_vector_type(8)))  float    v8f;
typedef __attribute__((ext_vector_type(4)))  unsigned v4u;
typedef __attribute__((ext_vector_type(8)))  int      v8i;
typedef __attribute__((ext_vector_type(4)))  int      v4i;

#define C_  32
#define D_  4096
#define N_  1024
#define B_  256
#define K_  21            // ceil(0.02 * 1024)
#define KSTEP 32          // WMMA K per step (f16)
#define NSTEPS (D_ / KSTEP)
#define NT  64            // N columns per workgroup
#define LDST 40           // padded LDS row stride (f16): 80B -> conflict-free b128 frags

#if __has_builtin(__builtin_amdgcn_tensor_load_to_lds)
#define HAVE_TDM 1
#else
#define HAVE_TDM 0
#endif

// ---------------- TDM: DMA one 32-row x 32-col f16 tile into LDS ----------------
// Tensor: f16 [rows][4096], tile start = gptr, row stride 4096 elems.
// LDS dest gets 4-DWORD pad every 16 DWORDs -> row stride 40 halfs == LDST.
__device__ __forceinline__ void tdm_x_tile(const _Float16* gptr, unsigned lds_byte_addr) {
#if HAVE_TDM
  unsigned long long ga = (unsigned long long)(uintptr_t)gptr;
  v4u g0;
  g0.x = 1u;                                                   // count=1 (valid), no gather
  g0.y = lds_byte_addr;                                        // lds_addr
  g0.z = (unsigned)ga;                                         // global_addr[31:0]
  g0.w = ((unsigned)(ga >> 32) & 0x01FFFFFFu) | (2u << 30);    // global_addr[56:32] | type=2
  v8i g1;
  g1[0] = (int)((1u << 16)      // data_size = 2B
              | (1u << 20)      // pad_enable
              | (3u << 22)      // pad_interval: 16 DWORDs (one 32-half row)
              | (3u << 25));    // pad_amount: 4 DWORDs (8 halfs) -> stride 40
  g1[1] = (int)((4096u & 0xFFFFu) << 16);                      // tensor_dim0 lo16 (=4096)
  g1[2] = (int)(((4096u >> 16) & 0xFFFFu) | ((32u) << 16));    // dim0 hi | tensor_dim1=32 lo
  g1[3] = (int)(32u << 16);                                    // dim1 hi=0 | tile_dim0=32
  g1[4] = (int)(32u);                                          // tile_dim1=32, tile_dim2=0
  g1[5] = (int)(4096u);                                        // tensor_dim0_stride lo32
  g1[6] = 0;                                                   // stride0 hi | stride1 lo
  g1[7] = 0;
  v4i z4 = {0, 0, 0, 0};
#if defined(__clang_major__) && __clang_major__ >= 23
  v8i z8 = {0, 0, 0, 0, 0, 0, 0, 0};
  __builtin_amdgcn_tensor_load_to_lds(g0, g1, z4, z4, z8, 0);
#else
  __builtin_amdgcn_tensor_load_to_lds(g0, g1, z4, z4, 0);
#endif
#else
  (void)gptr; (void)lds_byte_addr;
#endif
}

__device__ __forceinline__ void wait_tensorcnt0() {
#if __has_builtin(__builtin_amdgcn_s_wait_tensorcnt)
  __builtin_amdgcn_s_wait_tensorcnt(0);
#else
  asm volatile("s_wait_tensorcnt 0x0" ::: "memory");
#endif
}

// ---------------- LDS fragment loaders (row-major [row][k], stride LDST halfs) ----------------
// A 16x32 f16 (ISA 7.12.2): lanes 0-15 K{0-7,16-23}, lanes 16-31 K{8-15,24-31}
__device__ __forceinline__ v16h load_a_frag(const _Float16* base, int row, int lane) {
  const _Float16* p = base + (row + (lane & 15)) * LDST + ((lane >> 4) * 8);
  v8h lo = *(const v8h*)(p);
  v8h hi = *(const v8h*)(p + 16);
  v16h f;
#pragma unroll
  for (int i = 0; i < 8; ++i) { f[i] = lo[i]; f[8 + i] = hi[i]; }
  return f;
}
// B 32x16 f16 (stored transposed as [n][k]): lane%16 = col, lanes 0-15 K{0-7,16-23}, 16-31 K{8-15,24-31}
__device__ __forceinline__ v16h load_b_frag(const _Float16* base, int col, int lane) {
  const _Float16* p = base + (col + (lane & 15)) * LDST + ((lane >> 4) * 8);
  v8h lo = *(const v8h*)(p);
  v8h hi = *(const v8h*)(p + 16);
  v16h f;
#pragma unroll
  for (int i = 0; i < 8; ++i) { f[i] = lo[i]; f[8 + i] = hi[i]; }
  return f;
}

// ---------------- register-pipelined staging ----------------
struct CsRegs { float4 pm[2], pl[2]; };          // 32x64 connected slab / 256 thr = 8 elems
__device__ __forceinline__ void cs_load(const float* __restrict__ permC,
                                        const float* __restrict__ poolC,
                                        int d0, int tid, CsRegs& r) {
#pragma unroll
  for (int j = 0; j < 2; ++j) {
    int lin = j * 256 + tid;
    int k = lin >> 4, nc = (lin & 15) * 4;
    size_t g = (size_t)(d0 + k) * N_ + nc;
    r.pm[j] = *(const float4*)(permC + g);       // global_load_b128
    r.pl[j] = *(const float4*)(poolC + g);       // global_load_b128
  }
}
__device__ __forceinline__ void cs_store(_Float16* csb, int tid, const CsRegs& r) {
#pragma unroll
  for (int j = 0; j < 2; ++j) {
    int lin = j * 256 + tid;
    int k = lin >> 4, nc = (lin & 15) * 4;
    const float* pm = (const float*)&r.pm[j];
    const float* pl = (const float*)&r.pl[j];
#pragma unroll
    for (int i = 0; i < 4; ++i) {
      float cv = (pm[i] > 0.5f) ? pl[i] : 0.0f;
      csb[(nc + i) * LDST + k] = (_Float16)cv;   // transposed [n][k]
    }
  }
}
struct XRegs { float4 v[8]; };                   // fallback: 256x32 f32 slab / 256 thr
__device__ __forceinline__ void x_load(const float* __restrict__ x, int d0, int tid, XRegs& r) {
#pragma unroll
  for (int j = 0; j < 8; ++j) {
    int lin = j * 256 + tid;
    int m = lin >> 3, kc = (lin & 7) * 4;
    r.v[j] = *(const float4*)(x + (size_t)m * D_ + d0 + kc);
  }
}
__device__ __forceinline__ void x_store(_Float16* xsb, int tid, const XRegs& r) {
#pragma unroll
  for (int j = 0; j < 8; ++j) {
    int lin = j * 256 + tid;
    int m = lin >> 3, kc = (lin & 7) * 4;
    v4h h; h[0] = (_Float16)r.v[j].x; h[1] = (_Float16)r.v[j].y;
           h[2] = (_Float16)r.v[j].z; h[3] = (_Float16)r.v[j].w;
    *(v4h*)(xsb + m * LDST + kc) = h;            // ds_store_b64
  }
}

// ================= GEMM + boost kernel =================
// grid = (C, N/NT); block = 256 (8 waves). Wave (wm,wn): rows 64*wm..+63, cols 32*wn..+31.
template <bool USE_TDM>
__global__ __launch_bounds__(256) void pooler_gemm(
    const float*    __restrict__ x,     // f32 [B,D]       (fallback path)
    const _Float16* __restrict__ xh,    // f16 [B,D] in ws (TDM path)
    const float*    __restrict__ perm,  // [C,D,N]
    const float*    __restrict__ pool,  // [C,D,N]
    const float*    __restrict__ duty,  // [C,N]
    float*          __restrict__ out)   // [C,B,N]
{
  __shared__ _Float16 xs[2][B_ * LDST];   // x tile      [m][k], double buffered
  __shared__ _Float16 cs[2][NT * LDST];   // connected^T [n][k], double buffered

  const int tid  = threadIdx.x;
  const int lane = tid & 31;
  const int wave = tid >> 5;
  const int wm   = wave & 3;
  const int wn   = wave >> 2;
  const int c    = blockIdx.x;
  const int n0   = blockIdx.y * NT;

  const float* __restrict__ permC = perm + (size_t)c * D_ * N_ + n0;
  const float* __restrict__ poolC = pool + (size_t)c * D_ * N_ + n0;
  // per-wave TDM source/dest (each wave DMAs its 32 batch rows of the slab)
  const _Float16* xw = USE_TDM ? (xh + (size_t)(wave * 32) * D_) : nullptr;
  unsigned xlds[2];
  if (USE_TDM) {
    xlds[0] = (unsigned)(uintptr_t)(&xs[0][wave * 32 * LDST]);
    xlds[1] = (unsigned)(uintptr_t)(&xs[1][wave * 32 * LDST]);
  }

  const v8f vzero = {0.f, 0.f, 0.f, 0.f, 0.f, 0.f, 0.f, 0.f};
  v8f acc[4][2];
#pragma unroll
  for (int mt = 0; mt < 4; ++mt)
#pragma unroll
    for (int nt = 0; nt < 2; ++nt) acc[mt][nt] = vzero;

  // ---- prologue: fill buffer 0 ----
  CsRegs cr; XRegs xr;
  cs_load(permC, poolC, 0, tid, cr);
  if constexpr (!USE_TDM) x_load(x, 0, tid, xr);
  cs_store(cs[0], tid, cr);
  if constexpr (USE_TDM) { tdm_x_tile(xw, xlds[0]); wait_tensorcnt0(); }
  else                   { x_store(xs[0], tid, xr); }
  __syncthreads();

  for (int i = 0; i < NSTEPS; ++i) {
    const int cur = i & 1, nxt = cur ^ 1;
    const bool more = (i + 1) < NSTEPS;
    if (more) {
      const int d1 = (i + 1) * KSTEP;
      cs_load(permC, poolC, d1, tid, cr);               // long-latency, overlaps WMMA
      if constexpr (USE_TDM) tdm_x_tile(xw + d1, xlds[nxt]);  // async DMA into other buffer
      else                   x_load(x, d1, tid, xr);
    }
    // ---- WMMA phase on current buffers ----
    v16h bfrag[2];
#pragma unroll
    for (int nt = 0; nt < 2; ++nt)
      bfrag[nt] = load_b_frag(cs[cur], wn * 32 + nt * 16, lane);
#pragma unroll
    for (int mt = 0; mt < 4; ++mt) {
      v16h afrag = load_a_frag(xs[cur], wm * 64 + mt * 16, lane);
#pragma unroll
      for (int nt = 0; nt < 2; ++nt)
        acc[mt][nt] = __builtin_amdgcn_wmma_f32_16x16x32_f16(
            false, afrag, false, bfrag[nt], (short)0, acc[mt][nt], false, false);
    }
    if (more) {
      cs_store(cs[nxt], tid, cr);
      if constexpr (!USE_TDM) x_store(xs[nxt], tid, xr);
      if constexpr (USE_TDM) wait_tensorcnt0();
      __syncthreads();
    }
  }

  // ---- epilogue: fuse boost; C/D layout: col = lane%16, row = vgpr + 8*(lane/16) ----
#pragma unroll
  for (int nt = 0; nt < 2; ++nt) {
    const int n = n0 + wn * 32 + nt * 16 + (lane & 15);
    const float bst = __expf(100.0f * (0.02f - duty[c * N_ + n]));
#pragma unroll
    for (int mt = 0; mt < 4; ++mt) {
      const int mbase = wm * 64 + mt * 16 + ((lane >> 4) * 8);
#pragma unroll
      for (int i = 0; i < 8; ++i)
        out[((size_t)c * B_ + mbase + i) * N_ + n] = acc[mt][nt][i] * bst;
    }
  }
}

// ---------------- x f32 -> f16 pre-convert (feeds the TDM path) ----------------
__global__ __launch_bounds__(256) void xcvt_kernel(const float* __restrict__ x,
                                                   _Float16* __restrict__ xh) {
  const int i = blockIdx.x * 256 + threadIdx.x;     // one float4 per thread
  const float4 v = ((const float4*)x)[i];
  v4h h; h[0] = (_Float16)v.x; h[1] = (_Float16)v.y;
         h[2] = (_Float16)v.z; h[3] = (_Float16)v.w;
  ((v4h*)xh)[i] = h;
}

// ================= per-row top-K (radix select, K=21 of 1024) =================
__global__ __launch_bounds__(256) void topk_kernel(float* __restrict__ out)
{
  __shared__ unsigned hist[256];
  __shared__ unsigned sh_sel;
  __shared__ int      sh_k;

  const int tid = threadIdx.x;
  float* __restrict__ row = out + (size_t)blockIdx.x * N_;

  float v[4]; unsigned key[4];
#pragma unroll
  for (int j = 0; j < 4; ++j) { v[j] = row[tid + j * 256]; key[j] = __float_as_uint(v[j]); }

  unsigned prefix = 0;
  int kk = K_;
#pragma unroll
  for (int p = 24; p >= 0; p -= 8) {
    hist[tid] = 0;
    __syncthreads();
    const unsigned hm = (p == 24) ? 0u : (0xFFFFFFFFu << (p + 8));
#pragma unroll
    for (int j = 0; j < 4; ++j)
      if ((key[j] & hm) == (prefix & hm))
        atomicAdd(&hist[(key[j] >> p) & 0xFF], 1u);
    __syncthreads();
    if (tid == 0) {
      unsigned acc = 0;
      for (int b = 255; b >= 0; --b) {
        unsigned cb = hist[b];
        if (acc + cb >= (unsigned)kk) { sh_sel = (unsigned)b; sh_k = kk - (int)acc; break; }
        acc += cb;
      }
    }
    __syncthreads();
    prefix |= sh_sel << p;
    kk = sh_k;
  }

  const float kth = __uint_as_float(prefix);
#pragma unroll
  for (int j = 0; j < 4; ++j)
    row[tid + j * 256] = (v[j] >= kth) ? v[j] : 0.0f;
}

extern "C" void kernel_launch(void* const* d_in, const int* in_sizes, int n_in,
                              void* d_out, int out_size, void* d_ws, size_t ws_size,
                              hipStream_t stream) {
  (void)in_sizes; (void)n_in; (void)out_size;
  const float* x    = (const float*)d_in[0];
  const float* perm = (const float*)d_in[1];
  const float* pool = (const float*)d_in[2];
  const float* duty = (const float*)d_in[3];
  float* out = (float*)d_out;

  dim3 grid(C_, N_ / NT);
  const size_t need = (size_t)B_ * D_ * sizeof(_Float16);
#if HAVE_TDM
  if (ws_size >= need) {
    _Float16* xh = (_Float16*)d_ws;
    xcvt_kernel<<<(B_ * D_) / (256 * 4), 256, 0, stream>>>(x, xh);
    pooler_gemm<true><<<grid, 256, 0, stream>>>(x, xh, perm, pool, duty, out);
  } else
#endif
  {
    pooler_gemm<false><<<grid, 256, 0, stream>>>(x, nullptr, perm, pool, duty, out);
  }
  topk_kernel<<<C_ * B_, 256, 0, stream>>>(out);
}
